// AWD_LSTM_48447231099088
// MI455X (gfx1250) — compile-verified, matrix operand order checked by reading, not run on hardware
//
#include <hip/hip_runtime.h>
#include <hip/hip_bf16.h>

// AWD-LSTM forward for MI455X (gfx1250, wave32).
// bf16 WMMA (v_wmma_f32_16x16x32_bf16) for all GEMMs, f32 gates/cell state.
// Recurrent layers 1/2 use K-concatenated weights ([Wi|Wh], K=2048) so each
// LSTM cell needs a single GEMM; gate kernels write h directly into the packed
// activation buffers. GEMM inner loop is double-buffered with two disjoint
// register sets (ping-pong), 32-bit offsets off uniform bases.

typedef __bf16 bf16_t;
typedef __attribute__((ext_vector_type(16))) __bf16 v16bf;
typedef __attribute__((ext_vector_type(8)))  __bf16 v8bf;
typedef __attribute__((ext_vector_type(8)))  float  v8f;

#define SHUF16(lo, hi) __builtin_shufflevector((lo), (hi), 0,1,2,3,4,5,6,7,8,9,10,11,12,13,14,15)

namespace {
constexpr int kH = 1024;   // hidden = embedding
constexpr int kV = 32000;  // vocab
constexpr int kT = 70;     // sequence length
constexpr int kB = 64;     // batch
constexpr int kG = 4096;   // 4*H gate width
constexpr int kK2 = 2048;  // concatenated K for layers 1/2
}

__device__ __forceinline__ v8f wmma_bf16(v16bf a, v16bf b, v8f c) {
  return __builtin_amdgcn_wmma_f32_16x16x32_bf16(
      false, a, false, b, (short)0, c, false, false);
}

// ---------------- f32 -> bf16 bulk convert (n multiple of 4) ----------------
__global__ void cvt_f32_to_bf16(const float* __restrict__ src,
                                bf16_t* __restrict__ dst, int n) {
  int i = (blockIdx.x * blockDim.x + threadIdx.x) * 4;
  if (i >= n) return;
  float4 v = *(const float4*)(src + i);
  union { bf16_t h[4]; unsigned long long u; } u;
  u.h[0] = (bf16_t)v.x; u.h[1] = (bf16_t)v.y;
  u.h[2] = (bf16_t)v.z; u.h[3] = (bf16_t)v.w;
  *(unsigned long long*)(dst + i) = u.u;
}

// ------- concat two [rows,1024] f32 matrices into [rows,2048] bf16 ----------
__global__ void concat_cvt_bf16(const float* __restrict__ Wa,
                                const float* __restrict__ Wb,
                                bf16_t* __restrict__ dst, int rows) {
  int i = (blockIdx.x * blockDim.x + threadIdx.x) * 4;   // over rows*2048
  if (i >= rows * kK2) return;
  const int n = i >> 11;          // / 2048
  const int c = i & (kK2 - 1);
  const float* src = (c < kH) ? (Wa + (size_t)n * kH + c)
                              : (Wb + (size_t)n * kH + (c - kH));
  float4 v = *(const float4*)src;
  union { bf16_t h[4]; unsigned long long u; } u;
  u.h[0] = (bf16_t)v.x; u.h[1] = (bf16_t)v.y;
  u.h[2] = (bf16_t)v.z; u.h[3] = (bf16_t)v.w;
  *(unsigned long long*)(dst + i) = u.u;
}

// ------------------------------ bias combine --------------------------------
__global__ void add_bias_kernel(const float* __restrict__ a,
                                const float* __restrict__ b,
                                float* __restrict__ o, int n) {
  int i = blockIdx.x * blockDim.x + threadIdx.x;
  if (i < n) o[i] = a[i] + b[i];
}

// ------------- embedding gather + bf16 cast: one block per token ------------
__global__ void embed_gather_bf16(const int* __restrict__ x,
                                  const float* __restrict__ emb,
                                  bf16_t* __restrict__ xe) {
  const int row = blockIdx.x;                // t*B + b
  const int tok = x[row];
  const float* src = emb + (size_t)tok * kH;
  bf16_t* dst = xe + (size_t)row * kH;
  const int e = threadIdx.x * 4;             // blockDim 256 -> 1024 elems
  float4 v = *(const float4*)(src + e);
  union { bf16_t h[4]; unsigned long long u; } u;
  u.h[0] = (bf16_t)v.x; u.h[1] = (bf16_t)v.y;
  u.h[2] = (bf16_t)v.z; u.h[3] = (bf16_t)v.w;
  *(unsigned long long*)(dst + e) = u.u;
}

// ----------------------------- state init -----------------------------------
__global__ void init_state_kernel(const float* __restrict__ h0,
                                  const float* __restrict__ c0,
                                  bf16_t* __restrict__ h, int hStride,
                                  float* __restrict__ c, int layer) {
  int i = blockIdx.x * blockDim.x + threadIdx.x;
  if (i >= kB * kH) return;
  const int b = i >> 10, j = i & (kH - 1);
  size_t off = (size_t)layer * kB * kH + i;
  h[(size_t)b * hStride + j] = (bf16_t)h0[off];
  c[i] = c0[off];
}

// ---------- WMMA GEMM: C[M,N] = A[M,K](lda) * W[N,K]^T + bias ---------------
// A, W bf16; C f32 row-major (stride N). grid = (N/256, M/64), block = 256.
// One wave computes a 64x32 tile: 4 M-subtiles x 2 N-subtiles = 8 accumulators
// per k-step. Ping-pong double buffering: two disjoint fragment register sets,
// loads for step s+1 land while the 8 WMMAs for step s execute (no copies).
// Fragment layout per CDNA5 ISA (16-bit A 16x32 / B 32x16):
//   lane&15 selects row (A) / col (B); lane[4] owns K {8..15, 24..31}.
__global__ __launch_bounds__(256) void gemm_bf16_wmma(
    const bf16_t* __restrict__ A, int lda,
    const bf16_t* __restrict__ W,
    const float* __restrict__ bias, float* __restrict__ C, int N, int K) {
  const int lane = threadIdx.x & 31;
  const int wave = threadIdx.x >> 5;
  const int n0   = (blockIdx.x * 8 + wave) * 32;
  const int m0   = blockIdx.y * 64;
  const int r16  = lane & 15;
  const int hiK  = (lane >> 4) << 3;         // 0 or 8

  // 32-bit element offsets off uniform bases (saddr + voffset form)
  const int wOff0 = (n0 + r16) * K + hiK;
  const int wOff1 = wOff0 + 16 * K;
  const int aBase = (m0 + r16) * lda + hiK;
  const int aOff0 = aBase;
  const int aOff1 = aBase + 16 * lda;
  const int aOff2 = aBase + 32 * lda;
  const int aOff3 = aBase + 48 * lda;

  v8f acc[4][2] = {};

  v8bf w0lo[2], w0hi[2], w1lo[2], w1hi[2];
  v8bf aLo[2][4], aHi[2][4];

#define LOAD_SET(s, kk)                                                  \
  do {                                                                   \
    const int kbase = (kk);                                              \
    w0lo[s] = *(const v8bf*)(W + wOff0 + kbase);                         \
    w0hi[s] = *(const v8bf*)(W + wOff0 + kbase + 16);                    \
    w1lo[s] = *(const v8bf*)(W + wOff1 + kbase);                         \
    w1hi[s] = *(const v8bf*)(W + wOff1 + kbase + 16);                    \
    aLo[s][0] = *(const v8bf*)(A + aOff0 + kbase);                       \
    aHi[s][0] = *(const v8bf*)(A + aOff0 + kbase + 16);                  \
    aLo[s][1] = *(const v8bf*)(A + aOff1 + kbase);                       \
    aHi[s][1] = *(const v8bf*)(A + aOff1 + kbase + 16);                  \
    aLo[s][2] = *(const v8bf*)(A + aOff2 + kbase);                       \
    aHi[s][2] = *(const v8bf*)(A + aOff2 + kbase + 16);                  \
    aLo[s][3] = *(const v8bf*)(A + aOff3 + kbase);                       \
    aHi[s][3] = *(const v8bf*)(A + aOff3 + kbase + 16);                  \
  } while (0)

#define COMPUTE_SET(s)                                                   \
  do {                                                                   \
    v16bf B0 = SHUF16(w0lo[s], w0hi[s]);                                 \
    v16bf B1 = SHUF16(w1lo[s], w1hi[s]);                                 \
    v16bf F0 = SHUF16(aLo[s][0], aHi[s][0]);                             \
    v16bf F1 = SHUF16(aLo[s][1], aHi[s][1]);                             \
    v16bf F2 = SHUF16(aLo[s][2], aHi[s][2]);                             \
    v16bf F3 = SHUF16(aLo[s][3], aHi[s][3]);                             \
    acc[0][0] = wmma_bf16(F0, B0, acc[0][0]);                            \
    acc[0][1] = wmma_bf16(F0, B1, acc[0][1]);                            \
    acc[1][0] = wmma_bf16(F1, B0, acc[1][0]);                            \
    acc[1][1] = wmma_bf16(F1, B1, acc[1][1]);                            \
    acc[2][0] = wmma_bf16(F2, B0, acc[2][0]);                            \
    acc[2][1] = wmma_bf16(F2, B1, acc[2][1]);                            \
    acc[3][0] = wmma_bf16(F3, B0, acc[3][0]);                            \
    acc[3][1] = wmma_bf16(F3, B1, acc[3][1]);                            \
  } while (0)

  // K/32 is even for all call sites (K = 1024 or 2048).
  LOAD_SET(0, 0);
  int k = 32;
  const int pairs = (K >> 6) - 1;            // K/64 - 1 steady pairs
  for (int it = 0; it < pairs; ++it) {
    LOAD_SET(1, k);
    COMPUTE_SET(0);
    LOAD_SET(0, k + 32);
    COMPUTE_SET(1);
    k += 64;
  }
  LOAD_SET(1, k);
  COMPUTE_SET(0);
  COMPUTE_SET(1);

#undef LOAD_SET
#undef COMPUTE_SET

  // store: lane<16 -> (M=r, N=lane); lane>=16 -> (M=8+r, N=lane-16)
  const int rowOff = (lane >> 4) << 3;
#pragma unroll
  for (int nt = 0; nt < 2; ++nt) {
    const int cn = n0 + nt * 16 + r16;
    const float bv = bias ? bias[cn] : 0.0f;
#pragma unroll
    for (int mt = 0; mt < 4; ++mt) {
      float* cp = C + (size_t)(m0 + (mt << 4) + rowOff) * N + cn;
#pragma unroll
      for (int r = 0; r < 8; ++r)
        cp[(size_t)r * N] = acc[mt][nt][r] + bv;
    }
  }
}

// --------------------------- LSTM gate fusion --------------------------------
// p = (px ? px : 0) + pg  (biases folded into the GEMMs).
// Gate order per reference: i, f, o, g  (o = 3rd quarter, g = 4th).
__global__ void lstm_gate(const float* __restrict__ px,
                          const float* __restrict__ pg,
                          float* __restrict__ c,
                          bf16_t* __restrict__ h, int hStride,
                          bf16_t* __restrict__ h2, int h2Stride) {
  int i = blockIdx.x * blockDim.x + threadIdx.x;
  if (i >= kB * kH) return;
  const int b = i >> 10;
  const int j = i & (kH - 1);
  const float* qrow = pg + (size_t)b * kG;
  float pi = qrow[j];
  float pf = qrow[kH + j];
  float po = qrow[2 * kH + j];
  float pv = qrow[3 * kH + j];
  if (px) {
    const float* prow = px + (size_t)b * kG;
    pi += prow[j];
    pf += prow[kH + j];
    po += prow[2 * kH + j];
    pv += prow[3 * kH + j];
  }
  float ig = 1.0f / (1.0f + __expf(-pi));
  float fg = 1.0f / (1.0f + __expf(-pf));
  float og = 1.0f / (1.0f + __expf(-po));
  float g  = tanhf(pv);
  float c2 = fg * c[i] + ig * g;
  float h2v = og * tanhf(c2);
  c[i] = c2;
  bf16_t hv = (bf16_t)h2v;
  h[(size_t)b * hStride + j] = hv;
  if (h2) h2[(size_t)b * h2Stride + j] = hv;
}

// ------------------------------- driver --------------------------------------
extern "C" void kernel_launch(void* const* d_in, const int* in_sizes, int n_in,
                              void* d_out, int out_size, void* d_ws, size_t ws_size,
                              hipStream_t stream) {
  (void)in_sizes; (void)n_in; (void)out_size; (void)ws_size;
  const int*   x   = (const int*)  d_in[0];
  const float* h0  = (const float*)d_in[1];
  const float* c0  = (const float*)d_in[2];
  const float* emb = (const float*)d_in[3];
  const float* Wi0 = (const float*)d_in[4];
  const float* bi0 = (const float*)d_in[5];
  const float* Wh0 = (const float*)d_in[6];
  const float* bh0 = (const float*)d_in[7];
  const float* Wi1 = (const float*)d_in[8];
  const float* bi1 = (const float*)d_in[9];
  const float* Wh1 = (const float*)d_in[10];
  const float* bh1 = (const float*)d_in[11];
  const float* Wi2 = (const float*)d_in[12];
  const float* bi2 = (const float*)d_in[13];
  const float* Wh2 = (const float*)d_in[14];
  const float* bh2 = (const float*)d_in[15];
  const float* Wd  = (const float*)d_in[16];
  const float* bd  = (const float*)d_in[17];

  char* p = (char*)d_ws;
  auto take = [&p](size_t bytes) -> char* {
    char* r = p;
    p += (bytes + 255) & ~(size_t)255;
    return r;
  };

  const size_t wElems = (size_t)kG * kH;          // 4096*1024
  bf16_t* Wi0b = (bf16_t*)take(wElems * 2);
  bf16_t* Wh0b = (bf16_t*)take(wElems * 2);
  bf16_t* Wc1b = (bf16_t*)take((size_t)kG * kK2 * 2);   // [Wi1|Wh1]
  bf16_t* Wc2b = (bf16_t*)take((size_t)kG * kK2 * 2);   // [Wi2|Wh2]
  bf16_t* Wdb  = (bf16_t*)take((size_t)kV * kH * 2);
  float*  bc1  = (float*) take((size_t)kG * 4);         // bi1+bh1
  float*  bc2  = (float*) take((size_t)kG * 4);         // bi2+bh2
  bf16_t* xe   = (bf16_t*)take((size_t)kT * kB * kH * 2);
  float*  px0  = (float*) take((size_t)kT * kB * kG * 4);
  bf16_t* oseq = (bf16_t*)take((size_t)kT * kB * kH * 2);
  float*  pH   = (float*) take((size_t)kB * kG * 4);    // layer0 hidden proj
  float*  pC   = (float*) take((size_t)kB * kG * 4);    // layers 1/2 combined
  bf16_t* A1   = (bf16_t*)take((size_t)kB * kK2 * 2);   // [ha | hb_prev]
  bf16_t* A2   = (bf16_t*)take((size_t)kB * kK2 * 2);   // [hb | hc_prev]
  float*  ca   = (float*) take((size_t)kB * kH * 4);
  float*  cb   = (float*) take((size_t)kB * kH * 4);
  float*  cc   = (float*) take((size_t)kB * kH * 4);

  // 1) weight prep (bf16 copies / K-concat copies / combined biases)
  {
    int n = (int)wElems;
    cvt_f32_to_bf16<<<n / 1024, 256, 0, stream>>>(Wi0, Wi0b, n);
    cvt_f32_to_bf16<<<n / 1024, 256, 0, stream>>>(Wh0, Wh0b, n);
    int nc = kG * kK2;
    concat_cvt_bf16<<<nc / 1024, 256, 0, stream>>>(Wi1, Wh1, Wc1b, kG);
    concat_cvt_bf16<<<nc / 1024, 256, 0, stream>>>(Wi2, Wh2, Wc2b, kG);
    int nd = kV * kH;
    cvt_f32_to_bf16<<<nd / 1024, 256, 0, stream>>>(Wd, Wdb, nd);
    add_bias_kernel<<<kG / 256, 256, 0, stream>>>(bi1, bh1, bc1, kG);
    add_bias_kernel<<<kG / 256, 256, 0, stream>>>(bi2, bh2, bc2, kG);
  }

  // 2) embedding gather -> bf16
  embed_gather_bf16<<<kT * kB, 256, 0, stream>>>(x, emb, xe);

  // 3) initial states into packed slots
  {
    int blocks = (kB * kH) / 256;
    init_state_kernel<<<blocks, 256, 0, stream>>>(h0, c0, A1, kK2, ca, 0);       // ha
    init_state_kernel<<<blocks, 256, 0, stream>>>(h0, c0, A1 + kH, kK2, cb, 1);  // hb
    init_state_kernel<<<blocks, 256, 0, stream>>>(h0, c0, A2 + kH, kK2, cc, 2);  // hc
  }

  // 4) px0 = xe @ Wi0^T + bi0   (M=4480, N=4096, K=1024)
  gemm_bf16_wmma<<<dim3(kG / 256, (kT * kB) / 64), 256, 0, stream>>>(
      xe, kH, Wi0b, bi0, px0, kG, kH);

  // 5) recurrent scan: 3 GEMMs + 3 gate kernels per step
  const dim3 gStep(kG / 256, 1);
  const int gateBlocks = (kB * kH) / 256;
  for (int t = 0; t < kT; ++t) {
    const float* px_t = px0 + (size_t)t * kB * kG;
    // layer 0: pH = ha(t-1) @ Wh0^T + bh0 ; gates(px0[t] + pH) -> ha
    gemm_bf16_wmma<<<gStep, 256, 0, stream>>>(A1, kK2, Wh0b, bh0, pH, kG, kH);
    lstm_gate<<<gateBlocks, 256, 0, stream>>>(px_t, pH, ca, A1, kK2,
                                              nullptr, 0);
    // layer 1: pC = [ha | hb(t-1)] @ [Wi1|Wh1]^T + (bi1+bh1) ; gates -> hb
    //          hb written to A1[:,1024:] (for t+1) and A2[:,0:1024] (for cell2)
    gemm_bf16_wmma<<<gStep, 256, 0, stream>>>(A1, kK2, Wc1b, bc1, pC, kG, kK2);
    lstm_gate<<<gateBlocks, 256, 0, stream>>>(nullptr, pC, cb, A1 + kH, kK2,
                                              A2, kK2);
    // layer 2: pC = [hb | hc(t-1)] @ [Wi2|Wh2]^T + (bi2+bh2) ; gates -> hc
    //          hc written to A2[:,1024:] (for t+1) and oseq[t]
    gemm_bf16_wmma<<<gStep, 256, 0, stream>>>(A2, kK2, Wc2b, bc2, pC, kG, kK2);
    lstm_gate<<<gateBlocks, 256, 0, stream>>>(nullptr, pC, cc, A2 + kH, kK2,
                                              oseq + (size_t)t * kB * kH, kH);
  }

  // 6) decoder: d_out = oseq @ Wd^T + bd   (M=4480, N=32000, K=1024)
  gemm_bf16_wmma<<<dim3(kV / 256, (kT * kB) / 64), 256, 0, stream>>>(
      oseq, kH, Wdb, bd, (float*)d_out, kV, kH);
}
